// FeatureToTransformerPipeline_55009941127193
// MI455X (gfx1250) — matrix operand order repaired and checked by reading, not compiled, not run on hardware
//
#include <hip/hip_runtime.h>

typedef __attribute__((ext_vector_type(16))) _Float16 v16h;
typedef __attribute__((ext_vector_type(8)))  _Float16 v8h;
typedef __attribute__((ext_vector_type(8)))  float    v8f;

#define HIDDEN 768
#define NTOK   1024
#define BATCH  8
#define HEADS  12
#define HD     64
#define MTOT   (BATCH*NTOK)   /* 8192 */

// ---- CDNA5 async global->LDS copy (16B per lane, ASYNCcnt tracked) --------
__device__ __forceinline__ void async_cp16(unsigned lds_off, const void* gptr) {
  asm volatile("global_load_async_to_lds_b128 %0, %1, off"
               :: "v"(lds_off), "v"(gptr)
               : "memory");
}
__device__ __forceinline__ void wait_async0() {
  asm volatile("s_wait_asynccnt 0" ::: "memory");
}
#define LDSOFF(p) ((unsigned)(uintptr_t)(p))

// ---- WMMA fragment helpers (CDNA5 16x16x32 f16 layouts) -------------------
// A 16x32: lane m(<16) holds K={0..7} (elems 0..7) and K={16..23} (elems
// 8..15); lane m+16 holds K={8..15},{24..31}. p pre-offset by (lane>>4)*8.
__device__ __forceinline__ v16h frag_a32(const _Float16* p) {
  v8h lo = *(const v8h*)(p);
  v8h hi = *(const v8h*)(p + 16);
  v16h r;
#pragma unroll
  for (int i = 0; i < 8; ++i) { r[i] = lo[i]; r[i + 8] = hi[i]; }
  return r;
}
// B 32x16: lane n(<16) holds K={0..7},{16..23}; lane n+16 K={8..15},{24..31}. p pre-offset by (lane>>4)*8.
__device__ __forceinline__ v16h frag_b16(const _Float16* p) {
  v8h lo = *(const v8h*)(p);
  v8h hi = *(const v8h*)(p + 16);
  v16h r;
#pragma unroll
  for (int i = 0; i < 8; ++i) { r[i] = lo[i]; r[i + 8] = hi[i]; }
  return r;
}
__device__ __forceinline__ v8f wmma16(v16h a, v16h b, v8f c) {
  return __builtin_amdgcn_wmma_f32_16x16x32_f16(false, a, false, b,
                                                (short)0, c, false, false);
}
__device__ __forceinline__ void vzero(v8f& v) {
#pragma unroll
  for (int i = 0; i < 8; ++i) v[i] = 0.f;
}

// ---- Gaussian weight g[b,n] ----------------------------------------------
__global__ void g_kernel(const float* __restrict__ I, float* __restrict__ g) {
  int i = blockIdx.x * 256 + threadIdx.x;
  if (i < MTOT) {
    float x = I[i], x2 = x * x;
    float v = __expf(-x2 / (0.5f + 1e-5f)) +
              __expf(-x2 / (2.0f + 1e-5f)) +
              __expf(-x2 / (8.0f + 1e-5f));
    g[i] = v * (1.0f / 3.0f);
  }
}

// ---- One-time prep: feat[B,C,N] -> featT f16 [B*N][C] ---------------------
__global__ void cvt_featT_kernel(const float* __restrict__ feat,
                                 _Float16* __restrict__ dst) {
  int i = blockIdx.x * 256 + threadIdx.x;      // i = (b*256+c)*1024+n
  if (i < BATCH * 256 * 1024) {
    int n = i & 1023, c = (i >> 10) & 255, b = i >> 18;
    dst[(size_t)((b << 10) | n) * 256 + c] = (_Float16)feat[i];
  }
}
// ---- One-time prep: W f32 [K][N] -> Wt f16 [N][K] -------------------------
__global__ void cvt_trw_kernel(const float* __restrict__ src,
                               _Float16* __restrict__ dst, int K, int N) {
  int i = blockIdx.x * 256 + threadIdx.x;      // coalesced read over n
  if (i < K * N) {
    int n = i % N, k = i / N;
    dst[(size_t)n * K + k] = (_Float16)src[i];
  }
}

// ---- Unified double-buffered async GEMM: C = Ah[M][K] @ Bt[N][K]^T + bias -
// Block: 128 threads, tile 128M x 64N; wave owns 32 rows (2 A-frags,
// 8 accumulators -> 8 wmma per 32-wide K-step).
// MODE 0: f16 head-major [B,heads,N,64]  (q,k)
// MODE 1: f32 row-major                  (final output)
// MODE 2: f16 row-major + PE             (tokens)
// MODE 3: f16 transposed head-major [B,heads,64,N]  (v)
template <int MODE, int KDIM>
__global__ void gemm_kernel(const _Float16* __restrict__ Ah,
                            const _Float16* __restrict__ Bt,
                            const float* __restrict__ bias,
                            _Float16* __restrict__ out_h,
                            float* __restrict__ out_f) {
  __shared__ __align__(16) _Float16 la[2][128 * 32];  // 16 KB
  __shared__ __align__(16) _Float16 lb[2][64 * 32];   // 8 KB
  const int tid = threadIdx.x, lane = tid & 31, w = tid >> 5;
  const int m0 = blockIdx.x * 128, n0 = blockIdx.y * 64;
  const int ml = lane & 15, hi = lane >> 4, hi8 = hi * 8;

  v8f acc[2][4];
#pragma unroll
  for (int s = 0; s < 2; ++s)
#pragma unroll
    for (int t = 0; t < 4; ++t) vzero(acc[s][t]);

  // fixed-trip async tile copy (no exec-mask churn): A 512 chunks, B 256.
  auto issue_tile = [&](int bsel, int kt) {
#pragma unroll
    for (int it = 0; it < 4; ++it) {
      int idx = tid + it * 128;
      int m = idx >> 2, q = (idx & 3) * 8;
      async_cp16(LDSOFF(&la[bsel][m * 32 + q]),
                 Ah + (size_t)(m0 + m) * KDIM + kt + q);
    }
#pragma unroll
    for (int it = 0; it < 2; ++it) {
      int idx = tid + it * 128;
      int n = idx >> 2, q = (idx & 3) * 8;
      async_cp16(LDSOFF(&lb[bsel][n * 32 + q]),
                 Bt + (size_t)(n0 + n) * KDIM + kt + q);
    }
  };

  issue_tile(0, 0);
  int buf = 0;
  for (int kt = 0; kt < KDIM; kt += 32) {
    wait_async0();
    __syncthreads();
    if (kt + 32 < KDIM) issue_tile(buf ^ 1, kt + 32);
    v16h a0 = frag_a32(&la[buf][(32 * w + ml) * 32 + hi8]);
    v16h a1 = frag_a32(&la[buf][(32 * w + 16 + ml) * 32 + hi8]);
    v16h b[4];
#pragma unroll
    for (int t = 0; t < 4; ++t)
      b[t] = frag_b16(&lb[buf][(16 * t + ml) * 32 + hi8]);
#pragma unroll
    for (int t = 0; t < 4; ++t) acc[0][t] = wmma16(a0, b[t], acc[0][t]);
#pragma unroll
    for (int t = 0; t < 4; ++t) acc[1][t] = wmma16(a1, b[t], acc[1][t]);
    buf ^= 1;
  }

#pragma unroll
  for (int t = 0; t < 4; ++t) {
    int col = n0 + 16 * t + ml;
    float bv = bias[col];
    float dexp = 0.f;
    if (MODE == 2)
      dexp = __expf((float)(col & ~1) * (-9.2103403719761836f / 768.f));
#pragma unroll
    for (int s = 0; s < 2; ++s) {
#pragma unroll
      for (int r = 0; r < 8; ++r) {
        int row = m0 + 32 * w + 16 * s + r + hi8;
        float v = acc[s][t][r] + bv;
        if (MODE == 0) {
          int b_ = row >> 10, n_ = row & 1023;
          int head = col >> 6, dh = col & 63;
          out_h[(size_t)((b_ * HEADS + head) * NTOK + n_) * HD + dh] =
              (_Float16)v;
        } else if (MODE == 1) {
          out_f[(size_t)row * HIDDEN + col] = v;
        } else if (MODE == 2) {
          float ang = (float)(row & 1023) * dexp;
          v += (col & 1) ? __cosf(ang) : __sinf(ang);
          out_h[(size_t)row * HIDDEN + col] = (_Float16)v;
        } else {  // MODE 3: V transposed head-major
          int b_ = row >> 10, n_ = row & 1023;
          int head = col >> 6, dh = col & 63;
          out_h[(size_t)((b_ * HEADS + head) * HD + dh) * NTOK + n_] =
              (_Float16)v;
        }
      }
    }
  }
}

// ---- Flash attention: per (b,h,query-tile-64), async-streamed 64-key tiles
__global__ void attn_kernel(const _Float16* __restrict__ qh,
                            const _Float16* __restrict__ kh,
                            const _Float16* __restrict__ vth,
                            const float* __restrict__ g,
                            const float* __restrict__ lam_p,
                            _Float16* __restrict__ oh) {
  __shared__ __align__(16) _Float16 lq[64 * 64];      // [q][d]   8 KB
  __shared__ __align__(16) _Float16 lk[2][64 * 64];   // [k][d]  16 KB
  __shared__ __align__(16) _Float16 lvt[2][64 * 64];  // [d][k]  16 KB
  __shared__ __align__(16) _Float16 lp[4][16 * 64];   // P [q][k] 8 KB
  __shared__ float lg[NTOK];                          // 4 KB
  __shared__ float lsc[4][16];
  __shared__ float ll[4][16];

  const int tid = threadIdx.x, lane = tid & 31, w = tid >> 5;
  const int qt = blockIdx.x;           // 0..15
  const int bh = blockIdx.y;           // b*12+h
  const int b_ = bh / HEADS, h_ = bh % HEADS;
  const float lam = lam_p[0];
  const size_t base = (size_t)bh * NTOK * HD;   // q/k head-major base
  const size_t basev = (size_t)bh * HD * NTOK;  // v transposed base
  const int q0 = qt * 64;
  const int ml = lane & 15, hi = lane >> 4, hi8 = hi * 8;

  auto issue_kv = [&](int bsel, int k0) {
#pragma unroll
    for (int it = 0; it < 4; ++it) {
      int idx = tid + it * 128;
      int m = idx >> 3, q = (idx & 7) * 8;   // 64 rows x 64 halfs
      async_cp16(LDSOFF(&lk[bsel][m * 64 + q]),
                 kh + base + (size_t)(k0 + m) * HD + q);
      async_cp16(LDSOFF(&lvt[bsel][m * 64 + q]),
                 vth + basev + (size_t)m * NTOK + k0 + q);
    }
  };

  // prologue: async q tile + key/value tile 0; g row via VGPR path
#pragma unroll
  for (int it = 0; it < 4; ++it) {
    int idx = tid + it * 128;
    int m = idx >> 3, q = (idx & 7) * 8;
    async_cp16(LDSOFF(&lq[m * 64 + q]), qh + base + (size_t)(q0 + m) * HD + q);
  }
  issue_kv(0, 0);
#pragma unroll
  for (int it = 0; it < 8; ++it) {
    int idx = tid + it * 128;
    lg[idx] = g[b_ * NTOK + idx];
  }

  float m_run = -3.0e38f, l_run = 0.f;
  v8f o[4];
#pragma unroll
  for (int t = 0; t < 4; ++t) vzero(o[t]);

  int buf = 0;
  for (int k0 = 0; k0 < NTOK; k0 += 64) {
    wait_async0();
    __syncthreads();
    if (k0 + 64 < NTOK) issue_kv(buf ^ 1, k0 + 64);

    // S^T tiles: A = keys(16x32 over d), B = queries(cols); 4 key-subtiles
    v8f s[4];
#pragma unroll
    for (int st = 0; st < 4; ++st) vzero(s[st]);
    v16h bq0 = frag_b16(&lq[(16 * w + ml) * 64 + 0 + hi8]);
    v16h bq1 = frag_b16(&lq[(16 * w + ml) * 64 + 32 + hi8]);
#pragma unroll
    for (int st = 0; st < 4; ++st) {
      v16h a0 = frag_a32(&lk[buf][(16 * st + ml) * 64 + 0 + hi8]);
      v16h a1 = frag_a32(&lk[buf][(16 * st + ml) * 64 + 32 + hi8]);
      s[st] = wmma16(a0, bq0, s[st]);
      s[st] = wmma16(a1, bq1, s[st]);
    }

    // online softmax over keys: in-lane reduce + shfl_xor(16)
    float gql = lam * lg[q0 + 16 * w + ml];
    float vals[4][8];
    float tmax = -3.0e38f;
#pragma unroll
    for (int st = 0; st < 4; ++st)
#pragma unroll
      for (int r = 0; r < 8; ++r) {
        int kl = k0 + 16 * st + r + hi8;
        float v = s[st][r] * 0.125f + gql * lg[kl];
        vals[st][r] = v;
        tmax = fmaxf(tmax, v);
      }
    tmax = fmaxf(tmax, __shfl_xor(tmax, 16, 32));
    float m_new = fmaxf(m_run, tmax);
    float scale = __expf(m_run - m_new);
    float lsum = 0.f;
#pragma unroll
    for (int st = 0; st < 4; ++st) {
      v8h pv;
#pragma unroll
      for (int r = 0; r < 8; ++r) {
        float p = __expf(vals[st][r] - m_new);
        lsum += p;
        pv[r] = (_Float16)p;
      }
      *(v8h*)&lp[w][ml * 64 + 16 * st + hi8] = pv;  // P[q][k]
    }
    lsum += __shfl_xor(lsum, 16, 32);
    l_run = l_run * scale + lsum;
    m_run = m_new;
    if (lane < 16) lsc[w][lane] = scale;
    // lp/lsc are wave-private: same-wave DS ordering suffices, no barrier.

    // O update: O(16q x 64d) += P(16q x 64k) @ V(64k x 64d), 2 K-chunks
    v16h ap0 = frag_a32(&lp[w][ml * 64 + 0 + hi8]);
    v16h ap1 = frag_a32(&lp[w][ml * 64 + 32 + hi8]);
#pragma unroll
    for (int t = 0; t < 4; ++t) {
#pragma unroll
      for (int r = 0; r < 8; ++r) o[t][r] *= lsc[w][r + hi8];
      v16h bv0 = frag_b16(&lvt[buf][(16 * t + ml) * 64 + 0 + hi8]);
      v16h bv1 = frag_b16(&lvt[buf][(16 * t + ml) * 64 + 32 + hi8]);
      o[t] = wmma16(ap0, bv0, o[t]);
      o[t] = wmma16(ap1, bv1, o[t]);
    }
    buf ^= 1;
  }

  if (lane < 16) ll[w][lane] = l_run;
#pragma unroll
  for (int t = 0; t < 4; ++t) {
    int d = 16 * t + ml;
#pragma unroll
    for (int r = 0; r < 8; ++r) {
      int qrow = q0 + 16 * w + r + hi8;
      float val = o[t][r] / ll[w][r + hi8];
      oh[(size_t)(b_ * NTOK + qrow) * HIDDEN + h_ * HD + d] = (_Float16)val;
    }
  }
}

// ---------------------------------------------------------------------------
extern "C" void kernel_launch(void* const* d_in, const int* in_sizes, int n_in,
                              void* d_out, int out_size, void* d_ws,
                              size_t ws_size, hipStream_t stream) {
  const float* feat = (const float*)d_in[0];
  const float* I    = (const float*)d_in[1];
  const float* pw   = (const float*)d_in[2];
  const float* pb   = (const float*)d_in[3];
  const float* wq   = (const float*)d_in[4];
  const float* bq   = (const float*)d_in[5];
  const float* wk   = (const float*)d_in[6];
  const float* bk   = (const float*)d_in[7];
  const float* wv   = (const float*)d_in[8];
  const float* bv   = (const float*)d_in[9];
  const float* wo   = (const float*)d_in[10];
  const float* bo   = (const float*)d_in[11];
  const float* lam  = (const float*)d_in[12];

  char* ws = (char*)d_ws;
  float* g = (float*)ws;                              // 8192 f32
  _Float16* featT = (_Float16*)(ws + 32768);          // [8192][256]
  _Float16* pwt = featT + (size_t)MTOT * 256;         // [768][256]
  _Float16* wqt = pwt + (size_t)HIDDEN * 256;         // [768][768] x4
  _Float16* wkt = wqt + (size_t)HIDDEN * HIDDEN;
  _Float16* wvt = wkt + (size_t)HIDDEN * HIDDEN;
  _Float16* wot = wvt + (size_t)HIDDEN * HIDDEN;
  const size_t mat = (size_t)MTOT * HIDDEN;
  _Float16* tok = wot + (size_t)HIDDEN * HIDDEN;      // [8192][768]
  _Float16* qh  = tok + mat;                          // head-major
  _Float16* kh  = qh + mat;
  _Float16* vth = kh + mat;                           // transposed head-major
  _Float16* ah  = vth + mat;                          // attn out [8192][768]

  dim3 gblk(MTOT / 128, HIDDEN / 64), tb(128);

  g_kernel<<<dim3(32), dim3(256), 0, stream>>>(I, g);
  cvt_featT_kernel<<<dim3((BATCH * 256 * 1024) / 256), dim3(256), 0, stream>>>(
      feat, featT);
  cvt_trw_kernel<<<dim3((256 * HIDDEN + 255) / 256), dim3(256), 0, stream>>>(
      pw, pwt, 256, HIDDEN);
  cvt_trw_kernel<<<dim3((HIDDEN * HIDDEN + 255) / 256), dim3(256), 0, stream>>>(
      wq, wqt, HIDDEN, HIDDEN);
  cvt_trw_kernel<<<dim3((HIDDEN * HIDDEN + 255) / 256), dim3(256), 0, stream>>>(
      wk, wkt, HIDDEN, HIDDEN);
  cvt_trw_kernel<<<dim3((HIDDEN * HIDDEN + 255) / 256), dim3(256), 0, stream>>>(
      wv, wvt, HIDDEN, HIDDEN);
  cvt_trw_kernel<<<dim3((HIDDEN * HIDDEN + 255) / 256), dim3(256), 0, stream>>>(
      wo, wot, HIDDEN, HIDDEN);

  gemm_kernel<2, 256><<<gblk, tb, 0, stream>>>(featT, pwt, pb, tok, nullptr);
  gemm_kernel<0, 768><<<gblk, tb, 0, stream>>>(tok, wqt, bq, qh, nullptr);
  gemm_kernel<0, 768><<<gblk, tb, 0, stream>>>(tok, wkt, bk, kh, nullptr);
  gemm_kernel<3, 768><<<gblk, tb, 0, stream>>>(tok, wvt, bv, vth, nullptr);
  attn_kernel<<<dim3(NTOK / 64, BATCH * HEADS), tb, 0, stream>>>(
      qh, kh, vth, g, lam, ah);
  gemm_kernel<1, 768><<<gblk, tb, 0, stream>>>(ah, wot, bo, nullptr,
                                               (float*)d_out);
}
